// MultimodalEnergyAttention_8761733284253
// MI455X (gfx1250) — matrix operand order repaired and checked by reading, not compiled, not run on hardware
//
#include <hip/hip_runtime.h>
#include <hip/hip_bf16.h>
#include <math.h>

// ---------------------------------------------------------------------------
// Types for WMMA (gfx1250, wave32): v_wmma_f32_16x16x32_bf16
// ---------------------------------------------------------------------------
typedef __attribute__((ext_vector_type(16))) __bf16 v16bf;
typedef __attribute__((ext_vector_type(8)))  __bf16 v8bf;
typedef __attribute__((ext_vector_type(8)))  float  v8f;

__device__ inline unsigned short f2bf(float f) {
  unsigned int u = __float_as_uint(f);
  u += 0x7FFFu + ((u >> 16) & 1u);          // round-to-nearest-even
  return (unsigned short)(u >> 16);
}

// A-matrix fragment (16xK tile, K-chunk of 32, row-major bf16, stride lda).
// ISA layout: lane = m + 16*g ; v[0..7] = row[k0+8g .. +8), v[8..15] = row[k0+16+8g .. +8)
__device__ inline v16bf load_a_frag(const unsigned short* A, long lda, long rowBase,
                                    long k0, int lane) {
  int m = lane & 15, g = lane >> 4;
  const unsigned short* p = A + (rowBase + m) * lda + k0 + g * 8;
  v8bf lo = *reinterpret_cast<const v8bf*>(p);
  v8bf hi = *reinterpret_cast<const v8bf*>(p + 16);
  return __builtin_shufflevector(lo, hi, 0,1,2,3,4,5,6,7,8,9,10,11,12,13,14,15);
}

// B-matrix fragment (32x16 tile = K x N).  B[k][n] = W[n][k] with W row-major
// (N rows of K).  ISA layout: lane = n + 16*kg ; v[0..7] = k0+8kg .. +8, v[8..15] = k0+16+8kg .. +8 of col n.
__device__ inline v16bf load_b_frag(const unsigned short* W, long ldb, long colBase,
                                    long k0, int lane) {
  int n = lane & 15, kg = lane >> 4;
  const unsigned short* p = W + (colBase + n) * ldb + k0 + kg * 8;
  v8bf lo = *reinterpret_cast<const v8bf*>(p);
  v8bf hi = *reinterpret_cast<const v8bf*>(p + 16);
  return __builtin_shufflevector(lo, hi, 0,1,2,3,4,5,6,7,8,9,10,11,12,13,14,15);
}

__device__ inline v8f wmma_bf16(v16bf a, v16bf b, v8f c) {
  return __builtin_amdgcn_wmma_f32_16x16x32_bf16(false, a, false, b, (short)0, c,
                                                 false, false);
}

// ---------------------------------------------------------------------------
// Gauss-Legendre setup (50 points on (0,1)); writes weights + basis bin per point
// ---------------------------------------------------------------------------
__global__ void gl_setup_kernel(float* gw, int* bins) {
  int i = threadIdx.x;
  if (i >= 64) return;
  if (i < 50) {
    const double pi = 3.141592653589793238462643;
    const int n = 50;
    double x = cos(pi * (i + 0.75) / (n + 0.5));
    double p0 = 1.0, p1 = x, dp = 1.0;
    for (int it = 0; it < 64; ++it) {
      p0 = 1.0; p1 = x;
      for (int k = 2; k <= n; ++k) {
        double p2 = ((2.0 * k - 1.0) * x * p1 - (k - 1.0) * p0) / k;
        p0 = p1; p1 = p2;
      }
      dp = n * (x * p1 - p0) / (x * x - 1.0);
      x -= p1 / dp;
    }
    p0 = 1.0; p1 = x;
    for (int k = 2; k <= n; ++k) {
      double p2 = ((2.0 * k - 1.0) * x * p1 - (k - 1.0) * p0) / k;
      p0 = p1; p1 = p2;
    }
    dp = n * (x * p1 - p0) / (x * x - 1.0);
    double w = 2.0 / ((1.0 - x * x) * dp * dp);
    double t = 0.5 * (x + 1.0);
    gw[i] = (float)(0.5 * w);
    int bb = (int)floor(t * 128.0);
    bins[i] = bb < 0 ? 0 : (bb > 127 ? 127 : bb);
  } else {
    gw[i] = 0.f;
    bins[i] = 0;
  }
}

// ---------------------------------------------------------------------------
// fp32 -> bf16 conversion (grid-stride)
// ---------------------------------------------------------------------------
__global__ void cvt_kernel(const float* __restrict__ in,
                           unsigned short* __restrict__ out, long n) {
  for (long i = (long)blockIdx.x * blockDim.x + threadIdx.x; i < n;
       i += (long)gridDim.x * blockDim.x)
    out[i] = f2bf(in[i]);
}

// ---------------------------------------------------------------------------
// GEMM: C[M,768] = A[M,768] x W[768,768]^T  (bf16 in, bf16 out, fp32 acc)
// block = 256 thr (8 waves, 4 row-waves x 2 col-waves) -> 256 rows x 128 cols.
// wave tile = 64 rows x 64 cols = 4x4 WMMA tiles (16 WMMA / 8 frag loads).
// grid (M/256, 6)
// ---------------------------------------------------------------------------
__global__ __launch_bounds__(256) void gemm_store_kernel(
    const unsigned short* __restrict__ A, const unsigned short* __restrict__ W,
    unsigned short* __restrict__ C) {
  int wave = threadIdx.x >> 5, lane = threadIdx.x & 31;
  long rowBase = (long)blockIdx.x * 256 + (wave >> 1) * 64;
  int colBase = blockIdx.y * 128 + (wave & 1) * 64;
  v8f acc[4][4] = {};
  for (int k0 = 0; k0 < 768; k0 += 32) {
    v16bf a[4];
#pragma unroll
    for (int i = 0; i < 4; ++i)
      a[i] = load_a_frag(A, 768, rowBase + i * 16, k0, lane);
#pragma unroll
    for (int j = 0; j < 4; ++j) {
      v16bf bf = load_b_frag(W, 768, colBase + j * 16, k0, lane);
#pragma unroll
      for (int i = 0; i < 4; ++i)
        acc[i][j] = wmma_bf16(a[i], bf, acc[i][j]);
    }
  }
  int n = lane & 15, h2 = lane >> 4;
#pragma unroll
  for (int i = 0; i < 4; ++i)
#pragma unroll
    for (int j = 0; j < 4; ++j)
#pragma unroll
      for (int r = 0; r < 8; ++r) {
        long row = rowBase + i * 16 + r + 8 * h2;   // C/D layout: VGPR r, half h2
        C[row * 768 + colBase + j * 16 + n] = f2bf(acc[i][j][r]);
      }
}

// ---------------------------------------------------------------------------
// GEMM fused with rect-basis ridge compression:
// coeff[b,h,qk,m] = (sum_{l in bin m} (x W^T)[b*L+l, h*64+qk]) / (binSize+0.01)
// binSize = L/128; wave's 64 rows = 2 bins (text, 32/bin) or 4 bins (image).
// ---------------------------------------------------------------------------
__global__ __launch_bounds__(256) void gemm_compress_kernel(
    const unsigned short* __restrict__ A, const unsigned short* __restrict__ W,
    float* __restrict__ coeff, int L, int binSize) {
  int wave = threadIdx.x >> 5, lane = threadIdx.x & 31;
  long rowBase = (long)blockIdx.x * 256 + (wave >> 1) * 64;
  int colBase = blockIdx.y * 128 + (wave & 1) * 64;
  v8f acc[4][4] = {};
  for (int k0 = 0; k0 < 768; k0 += 32) {
    v16bf a[4];
#pragma unroll
    for (int i = 0; i < 4; ++i)
      a[i] = load_a_frag(A, 768, rowBase + i * 16, k0, lane);
#pragma unroll
    for (int j = 0; j < 4; ++j) {
      v16bf bf = load_b_frag(W, 768, colBase + j * 16, k0, lane);
#pragma unroll
      for (int i = 0; i < 4; ++i)
        acc[i][j] = wmma_bf16(a[i], bf, acc[i][j]);
    }
  }
  int b = (int)(rowBase / L);
  int lrow = (int)(rowBase % L);
  float inv = 1.0f / ((float)binSize + 0.01f);
#pragma unroll
  for (int j = 0; j < 4; ++j) {
    float fs[4];
#pragma unroll
    for (int i = 0; i < 4; ++i) {
      float cs = 0.f;
#pragma unroll
      for (int r = 0; r < 8; ++r) cs += acc[i][j][r];     // rows of one half
      cs += __shfl_xor(cs, 16, 32);                       // join halves -> 16 rows
      fs[i] = cs;
    }
    if (lane < 16) {
      int c = colBase + j * 16 + lane;                    // c = h*64 + qk
      int h = c >> 6, qk = c & 63;
      long basei = ((long)(b * 12 + h) * 64 + qk) * 128;
      if (binSize == 16) {
#pragma unroll
        for (int i = 0; i < 4; ++i)
          coeff[basei + (lrow >> 4) + i] = fs[i] * inv;
      } else {
        coeff[basei + (lrow >> 5)]     = (fs[0] + fs[1]) * inv;
        coeff[basei + (lrow >> 5) + 1] = (fs[2] + fs[3]) * inv;
      }
    }
  }
}

// ---------------------------------------------------------------------------
// Build transposed reconstruction tiles: Rt[b,h,t,q] = coeff[b,h,q,bin(t)]
// (t padded to 64, zeros for t>=50).  bf16 so score GEMM B-frags are contiguous.
// ---------------------------------------------------------------------------
__global__ void rt_build_kernel(const float* __restrict__ coeff,
                                const int* __restrict__ bins,
                                unsigned short* __restrict__ Rt) {
  long idx = (long)blockIdx.x * blockDim.x + threadIdx.x;   // 4*12*64*64 threads
  if (idx >= (long)4 * 12 * 64 * 64) return;
  int q  = (int)(idx & 63);
  int t  = (int)((idx >> 6) & 63);
  long bh = idx >> 12;
  float v = 0.f;
  if (t < 50) v = coeff[(bh * 64 + q) * 128 + bins[t]];
  Rt[(bh * 64 + t) * 64 + q] = f2bf(v);
}

// ---------------------------------------------------------------------------
// Scores + logsumexp:  S[l,t] = sum_q Q[b*L+l, h*64+q] * Rt[b,h,t,q]
// then per-row: log( sum_t exp(BETA*(S-m)) * w_t + 1e-15 ) + BETA*m
// grid (L/128, 12, 4); wave -> 16 rows x 64 t-cols (K=64).
// ---------------------------------------------------------------------------
__global__ __launch_bounds__(256) void score_kernel(
    const unsigned short* __restrict__ Q, const unsigned short* __restrict__ Rt,
    const float* __restrict__ gw, int L, float* __restrict__ partials, int pbase) {
  int wave = threadIdx.x >> 5, lane = threadIdx.x & 31;
  int b = blockIdx.z, h = blockIdx.y;
  long rowBase = (long)b * L + (long)blockIdx.x * 128 + wave * 16;
  const unsigned short* Bp = Rt + ((long)(b * 12 + h) * 64) * 64;
  v8f acc[4] = {};
#pragma unroll
  for (int kk = 0; kk < 64; kk += 32) {
    v16bf a = load_a_frag(Q, 768, rowBase, h * 64 + kk, lane);
#pragma unroll
    for (int j = 0; j < 4; ++j) {
      v16bf bf = load_b_frag(Bp, 64, j * 16, kk, lane);
      acc[j] = wmma_bf16(a, bf, acc[j]);
    }
  }
  __shared__ float S[8][16][65];   // stride 65 -> conflict-free row reads
  __shared__ float red[256];
  int n = lane & 15, h2 = lane >> 4;
#pragma unroll
  for (int j = 0; j < 4; ++j)
#pragma unroll
    for (int r = 0; r < 8; ++r)
      S[wave][r + 8 * h2][j * 16 + n] = acc[j][r];
  __syncthreads();
  float rowres = 0.f;
  if (lane < 16) {
    float m = -3.4e38f;
    for (int t = 0; t < 50; ++t) m = fmaxf(m, S[wave][lane][t]);
    float ssum = 0.f;
    for (int t = 0; t < 50; ++t)
      ssum += __expf(0.125f * (S[wave][lane][t] - m)) * gw[t];
    rowres = __logf(ssum + 1e-15f) + 0.125f * m;
  }
  red[threadIdx.x] = rowres;
  __syncthreads();
  if (threadIdx.x == 0) {
    float s = 0.f;
    for (int i = 0; i < 256; ++i) s += red[i];   // fixed order -> deterministic
    partials[pbase + ((blockIdx.z * gridDim.y + blockIdx.y) * gridDim.x + blockIdx.x)] = s;
  }
}

// ---------------------------------------------------------------------------
// Sum of squares partials (fixed grid, deterministic tree reduction)
// ---------------------------------------------------------------------------
__global__ void sq_kernel(const float* __restrict__ x, long n,
                          float* __restrict__ partials) {
  __shared__ float red[256];
  float s = 0.f;
  for (long i = (long)blockIdx.x * blockDim.x + threadIdx.x; i < n;
       i += (long)gridDim.x * blockDim.x) {
    float v = x[i];
    s += v * v;
  }
  red[threadIdx.x] = s;
  __syncthreads();
  for (int st = 128; st > 0; st >>= 1) {
    if ((int)threadIdx.x < st) red[threadIdx.x] += red[threadIdx.x + st];
    __syncthreads();
  }
  if (threadIdx.x == 0) partials[blockIdx.x] = red[0];
}

// ---------------------------------------------------------------------------
// Final: out = -(1/BETA)*sum(logz partials) + 0.5*sum(sq partials);  BETA=0.125
// ---------------------------------------------------------------------------
__global__ void final_reduce_kernel(const float* __restrict__ plz, int nlz,
                                    const float* __restrict__ psq, int nsq,
                                    float* __restrict__ out) {
  __shared__ float r1[256], r2[256];
  float a = 0.f, b = 0.f;
  for (int i = threadIdx.x; i < nlz; i += 256) a += plz[i];
  for (int i = threadIdx.x; i < nsq; i += 256) b += psq[i];
  r1[threadIdx.x] = a; r2[threadIdx.x] = b;
  __syncthreads();
  for (int st = 128; st > 0; st >>= 1) {
    if ((int)threadIdx.x < st) {
      r1[threadIdx.x] += r1[threadIdx.x + st];
      r2[threadIdx.x] += r2[threadIdx.x + st];
    }
    __syncthreads();
  }
  if (threadIdx.x == 0) out[0] = -8.0f * r1[0] + 0.5f * r2[0];
}

// ---------------------------------------------------------------------------
extern "C" void kernel_launch(void* const* d_in, const int* in_sizes, int n_in,
                              void* d_out, int out_size, void* d_ws, size_t ws_size,
                              hipStream_t stream) {
  (void)in_sizes; (void)n_in; (void)out_size; (void)ws_size;
  const float* x_text  = (const float*)d_in[0];
  const float* x_image = (const float*)d_in[1];
  const float* wsrc[6] = {(const float*)d_in[2], (const float*)d_in[3],
                          (const float*)d_in[4], (const float*)d_in[5],
                          (const float*)d_in[6], (const float*)d_in[7]};
  float* out = (float*)d_out;

  const long SZ_XT = (long)4 * 4096 * 768;   // 12,582,912
  const long SZ_XI = (long)4 * 2048 * 768;   //  6,291,456
  const long SZ_W  = (long)768 * 768;

  char* base = (char*)d_ws;
  size_t off = 0;
  auto take = [&](size_t bytes) -> char* {
    char* p = base + off;
    off += (bytes + 255) & ~(size_t)255;
    return p;
  };
  float* gw   = (float*)take(64 * sizeof(float));
  int*   bins = (int*)take(64 * sizeof(int));
  unsigned short* xt_bf = (unsigned short*)take(SZ_XT * 2);
  unsigned short* xi_bf = (unsigned short*)take(SZ_XI * 2);
  unsigned short* wbf[6];
  for (int i = 0; i < 6; ++i) wbf[i] = (unsigned short*)take(SZ_W * 2);
  unsigned short* bufA = (unsigned short*)take(SZ_XT * 2);  // Q_text, then Xp_text
  unsigned short* bufB = (unsigned short*)take(SZ_XT * 2);  // Qc text->image
  unsigned short* bufC = (unsigned short*)take(SZ_XI * 2);  // Q_image, then Xp_image
  unsigned short* bufD = (unsigned short*)take(SZ_XI * 2);  // Qc image->text
  float* coeff_t = (float*)take((size_t)4 * 12 * 64 * 128 * 4);
  float* coeff_i = (float*)take((size_t)4 * 12 * 64 * 128 * 4);
  unsigned short* rt_t = (unsigned short*)take((size_t)4 * 12 * 64 * 64 * 2);
  unsigned short* rt_i = (unsigned short*)take((size_t)4 * 12 * 64 * 64 * 2);
  float* p_logz = (float*)take(4608 * sizeof(float));
  float* p_sq   = (float*)take(1536 * sizeof(float));

  // wbf index map: 0=wq_text 1=wk_text 2=wq_image 3=wk_image 4=w_t2i 5=w_i2t
  gl_setup_kernel<<<1, 64, 0, stream>>>(gw, bins);
  cvt_kernel<<<2048, 256, 0, stream>>>(x_text, xt_bf, SZ_XT);
  cvt_kernel<<<1024, 256, 0, stream>>>(x_image, xi_bf, SZ_XI);
  for (int i = 0; i < 6; ++i)
    cvt_kernel<<<512, 256, 0, stream>>>(wsrc[i], wbf[i], SZ_W);

  // K compressions (fused GEMM + bin-sum)
  gemm_compress_kernel<<<dim3(64, 6), 256, 0, stream>>>(xt_bf, wbf[1], coeff_t, 4096, 32);
  gemm_compress_kernel<<<dim3(32, 6), 256, 0, stream>>>(xi_bf, wbf[3], coeff_i, 2048, 16);
  rt_build_kernel<<<768, 256, 0, stream>>>(coeff_t, bins, rt_t);
  rt_build_kernel<<<768, 256, 0, stream>>>(coeff_i, bins, rt_i);

  // intra text
  gemm_store_kernel<<<dim3(64, 6), 256, 0, stream>>>(xt_bf, wbf[0], bufA);
  score_kernel<<<dim3(32, 12, 4), 256, 0, stream>>>(bufA, rt_t, gw, 4096, p_logz, 0);
  // cross text->image  (q from projected text, k = image)
  gemm_store_kernel<<<dim3(64, 6), 256, 0, stream>>>(xt_bf, wbf[4], bufA);  // x_proj
  gemm_store_kernel<<<dim3(64, 6), 256, 0, stream>>>(bufA, wbf[2], bufB);   // q
  score_kernel<<<dim3(32, 12, 4), 256, 0, stream>>>(bufB, rt_i, gw, 4096, p_logz, 1536);
  // intra image
  gemm_store_kernel<<<dim3(32, 6), 256, 0, stream>>>(xi_bf, wbf[2], bufC);
  score_kernel<<<dim3(16, 12, 4), 256, 0, stream>>>(bufC, rt_i, gw, 2048, p_logz, 3072);
  // cross image->text
  gemm_store_kernel<<<dim3(32, 6), 256, 0, stream>>>(xi_bf, wbf[5], bufC);  // x_proj
  gemm_store_kernel<<<dim3(32, 6), 256, 0, stream>>>(bufC, wbf[0], bufD);   // q
  score_kernel<<<dim3(16, 12, 4), 256, 0, stream>>>(bufD, rt_t, gw, 2048, p_logz, 3840);

  // 0.5*||x||^2 terms
  sq_kernel<<<1024, 256, 0, stream>>>(x_text, SZ_XT, p_sq);
  sq_kernel<<<512, 256, 0, stream>>>(x_image, SZ_XI, p_sq + 1024);

  final_reduce_kernel<<<1, 256, 0, stream>>>(p_logz, 4608, p_sq, 1536, out);
}
